// MHSAttention_23252952940835
// MI455X (gfx1250) — compile-verified, matrix operand order checked by reading, not run on hardware
//
#include <hip/hip_runtime.h>

// MHSA for MI455X (gfx1250): all matmuls via v_wmma_f32_16x16x32_bf16,
// double-buffered async global->LDS staging (ASYNCcnt) overlapping DMA with
// WMMA compute, fp32 softmax/RoPE. Wave32 fragment layouts per ISA 7.12.2.

typedef __bf16 bf16;
typedef bf16  v8bf  __attribute__((ext_vector_type(8)));
typedef bf16  v16bf __attribute__((ext_vector_type(16)));
typedef float v8f   __attribute__((ext_vector_type(8)));

static constexpr int B_  = 2;
static constexpr int S_  = 2048;
static constexpr int H_  = 2048;
static constexpr int NH_ = 16;
static constexpr int HD_ = 128;
static constexpr float SCALE_ = 0.08838834764831845f; // 1/sqrt(128)

// ---- fragment helpers -------------------------------------------------------
// A-frag (16x32 bf16): lane L holds row M=L%16, K-chunks {hf*8..} and {16+hf*8..}.
// B-frag (32x16) mirrors with lane = column N, K-contiguous.
__device__ __forceinline__ v16bf frag_ld(const bf16* base, int stride, int row,
                                         int k0, int hf) {
  const bf16* p = base + (size_t)row * stride + k0 + hf * 8;
  v8bf lo = *(const v8bf*)p;
  v8bf hi = *(const v8bf*)(p + 16);
  return __builtin_shufflevector(lo, hi, 0,1,2,3,4,5,6,7,8,9,10,11,12,13,14,15);
}

__device__ __forceinline__ v8f wmma_bf16(v16bf a, v16bf b, v8f c) {
  return __builtin_amdgcn_wmma_f32_16x16x32_bf16(false, a, false, b,
                                                 (short)0, c, false, false);
}

// async global->LDS b128 (GVS mode): lds byte addr (VGPR), 32-bit byte voffset
// (VGPR), 64-bit base (SGPR pair). Tracked by ASYNCcnt.
__device__ __forceinline__ void async_ld_b128(unsigned lds_addr, unsigned voff,
                                              const bf16* sbase) {
  asm volatile("global_load_async_to_lds_b128 %0, %1, %2"
               :: "v"(lds_addr), "v"(voff), "s"(sbase) : "memory");
}
__device__ __forceinline__ void wait_async0() {
  asm volatile("s_wait_asynccnt 0x0" ::: "memory");
}
__device__ __forceinline__ unsigned lds_off_u32(const void* p) {
  return (unsigned)(uintptr_t)p;   // AS3->flat cast keeps LDS offset in low 32b
}

// ---- converters -------------------------------------------------------------
__global__ void cvt_bf16(const float* __restrict__ in, bf16* __restrict__ out, int n) {
  int i = blockIdx.x * blockDim.x + threadIdx.x;
  if (i < n) out[i] = (bf16)in[i];
}

// in[R][C] fp32 -> out[C][R] bf16
__global__ void cvt_bf16_t(const float* __restrict__ in, bf16* __restrict__ out,
                           int R, int C) {
  int i = blockIdx.x * blockDim.x + threadIdx.x;
  if (i < R * C) {
    int r = i / C, c = i % C;
    out[(size_t)c * R + r] = (bf16)in[i];
  }
}

// ---- fused QKV projection + bias + RoPE ------------------------------------
// grid (B*S/64, 8, 3), block 256 (8 waves). Block double-buffers a 64x32 A
// tile in LDS via async DMA (overlapped with compute); each wave owns one
// (head, d-block) pair of columns (d and d+64 -> rotate_half in-register)
// across the block's 4 M-tiles: 8 WMMAs per K-step.
__global__ __launch_bounds__(256) void qkv_rope_kernel(
    const bf16* __restrict__ hid,          // [B*S][H] bf16
    const bf16* __restrict__ Wqt, const bf16* __restrict__ Wkt,
    const bf16* __restrict__ Wvt,          // each [NH*HD][H] (transposed)
    const float* __restrict__ bq, const float* __restrict__ bk,
    const float* __restrict__ bv,
    bf16* __restrict__ Q, bf16* __restrict__ K,  // [B][NH][S][HD]
    bf16* __restrict__ Vt)                        // [B][NH][HD][S]
{
  __shared__ bf16 Alds[2][64 * 32];              // 2 x 4 KB A tiles

  const int which = blockIdx.z;
  const bf16*  Wt   = which == 0 ? Wqt : which == 1 ? Wkt : Wvt;
  const float* bias = which == 0 ? bq  : which == 1 ? bk  : bv;

  const int tid  = threadIdx.x;
  const int wave = tid >> 5;
  const int lane = tid & 31;
  const int hf   = lane >> 4;
  const int l16  = lane & 15;

  const int row0b = blockIdx.x * 64;             // block's 64 rows of [B*S]
  const int combo = blockIdx.y * 8 + wave;       // 0..63
  const int head  = combo >> 2;
  const int db    = combo & 3;                   // d block in [0,64)
  const int col0  = head * HD_ + db * 16;
  const int col1  = col0 + 64;

  // staging: one b128 chunk per thread per K-step (64 rows x 32 k = 256 chunks)
  const int srow = tid >> 2;                     // 0..63
  const int sk16 = tid & 3;                      // 16B sub-chunk within 64B row
  const unsigned ldsA0 = lds_off_u32(&Alds[0][0]) + tid * 16;
  const unsigned ldsA1 = lds_off_u32(&Alds[1][0]) + tid * 16;
  const bf16*    Abase = hid + (size_t)row0b * H_;
  const unsigned vbase = (unsigned)(srow * (H_ * 2) + sk16 * 16);

  v8f c[8];
#pragma unroll
  for (int i = 0; i < 8; ++i) c[i] = (v8f){};

  async_ld_b128(ldsA0, vbase, Abase);            // prologue: tile 0
  int ibuf = 0;
  for (int k0 = 0; k0 < H_; k0 += 32) {
    wait_async0();                               // tile k0 landed
    __syncthreads();                             // ...everywhere; prev compute done
    if (k0 + 32 < H_)                            // stage next tile (overlapped)
      async_ld_b128(ibuf ? ldsA0 : ldsA1, vbase + (unsigned)((k0 + 32) * 2), Abase);

    v16bf b0 = frag_ld(Wt, H_, col0 + l16, k0, hf);
    v16bf b1 = frag_ld(Wt, H_, col1 + l16, k0, hf);
    __builtin_prefetch(Wt + (size_t)(col0 + l16) * H_ + k0 + 64, 0, 1);
    const bf16* Ab = &Alds[ibuf][0];
#pragma unroll
    for (int mt = 0; mt < 4; ++mt) {
      v16bf a = frag_ld(Ab, 32, mt * 16 + l16, 0, hf);
      c[mt * 2]     = wmma_bf16(a, b0, c[mt * 2]);
      c[mt * 2 + 1] = wmma_bf16(a, b1, c[mt * 2 + 1]);
    }
    ibuf ^= 1;
  }

  const float bias0 = bias[col0 + l16];
  const float bias1 = bias[col1 + l16];
  const int   d     = db * 16 + l16;             // [0,64)
  const float invf  = __powf(10000.0f, -(float)d * (2.0f / (float)HD_));

#pragma unroll
  for (int mt = 0; mt < 4; ++mt) {
#pragma unroll
    for (int i = 0; i < 8; ++i) {
      const int grow = row0b + mt * 16 + i + hf * 8;
      const int b = grow / S_;
      const int s = grow % S_;
      float x0 = c[mt * 2][i] + bias0;
      float x1 = c[mt * 2 + 1][i] + bias1;
      if (which < 2) {                           // RoPE for q,k
        const float ang = (float)s * invf;
        const float cs = __cosf(ang), sn = __sinf(ang);
        const float r0 = x0 * cs - x1 * sn;
        const float r1 = x1 * cs + x0 * sn;
        x0 = r0; x1 = r1;
      }
      if (which < 2) {
        bf16* dst = (which == 0) ? Q : K;
        const size_t base = (((size_t)b * NH_ + head) * S_ + s) * HD_;
        dst[base + d]      = (bf16)x0;
        dst[base + d + 64] = (bf16)x1;
      } else {
        const size_t base = ((size_t)b * NH_ + head) * HD_;
        Vt[(base + d)      * S_ + s] = (bf16)x0;
        Vt[(base + d + 64) * S_ + s] = (bf16)x1;
      }
    }
  }
}

// ---- flash attention (causal, online softmax) ------------------------------
// grid (S/128, B*NH), block 256. Block double-buffers shared 32x128 K and
// 128x32 V^T tiles via async DMA (overlapped with compute); each wave owns a
// 16-row q strip with fp32 online softmax.
__global__ __launch_bounds__(256) void flash_attn_kernel(
    const bf16* __restrict__ Q,   // [B][NH][S][HD]
    const bf16* __restrict__ K,   // [B][NH][S][HD]
    const bf16* __restrict__ Vt,  // [B][NH][HD][S]
    bf16* __restrict__ out)       // [B*S][NH*HD]
{
  __shared__ bf16 Klds[2][32 * 128];             // 2 x 8 KB  [k_local][d]
  __shared__ bf16 Vlds[2][128 * 32];             // 2 x 8 KB  [d][s_local]
  __shared__ bf16 plds[8][16][32];               // 8 KB per-wave P staging

  const int tid  = threadIdx.x;
  const int wave = tid >> 5;
  const int lane = tid & 31;
  const int hf   = lane >> 4;
  const int l16  = lane & 15;

  const int bn   = blockIdx.y;
  const int b    = bn / NH_;
  const int head = bn % NH_;
  const int q0b  = blockIdx.x * 128;
  const int q0   = q0b + wave * 16;

  const bf16* Qp = Q  + (size_t)bn * S_ * HD_;
  const bf16* Kp = K  + (size_t)bn * S_ * HD_;
  const bf16* Vp = Vt + (size_t)bn * HD_ * S_;

  v16bf qf[4];
#pragma unroll
  for (int dc = 0; dc < 4; ++dc)
    qf[dc] = frag_ld(Qp, HD_, q0 + l16, dc * 32, hf);

  float mrow[8], lrow[8];
  v8f o[8];
#pragma unroll
  for (int i = 0; i < 8; ++i) { mrow[i] = -1e30f; lrow[i] = 0.0f; }
#pragma unroll
  for (int dc = 0; dc < 8; ++dc) o[dc] = (v8f){};

  const unsigned ldsK[2] = { lds_off_u32(&Klds[0][0]), lds_off_u32(&Klds[1][0]) };
  const unsigned ldsV[2] = { lds_off_u32(&Vlds[0][0]), lds_off_u32(&Vlds[1][0]) };

  const int kend = q0 + 16;                      // this wave's causal bound
  const int kmax = q0b + 128;                    // block's shared k range

  // cooperative async staging: 512 b128 chunks each for K and V tiles
  auto stage = [&](int kb, int ib) {
#pragma unroll
    for (int j = 0; j < 2; ++j) {
      const int cc = tid + j * 256;
      const int kl = cc >> 4, k16 = cc & 15;     // K: [kl][k16*8 d-chunk]
      async_ld_b128(ldsK[ib] + cc * 16,
                    (unsigned)((kb + kl) * (HD_ * 2) + k16 * 16), Kp);
      const int dr = cc >> 2, s4 = cc & 3;       // V: [dr][s4*8 s-chunk]
      async_ld_b128(ldsV[ib] + cc * 16,
                    (unsigned)(dr * (S_ * 2) + kb * 2 + s4 * 16), Vp);
    }
  };

  stage(0, 0);                                   // prologue: tile 0
  int ibuf = 0;
  for (int kb = 0; kb < kmax; kb += 32) {
    wait_async0();                               // tile kb landed
    __syncthreads();                             // ...everywhere; prev compute done
    if (kb + 32 < kmax) stage(kb + 32, ibuf ^ 1);

    if (kb < kend) {
      const bf16* Kb = &Klds[ibuf][0];
      const bf16* Vb = &Vlds[ibuf][0];
      v8f s0 = {}; v8f s1 = {};
#pragma unroll
      for (int dc = 0; dc < 4; ++dc) {
        v16bf kf0 = frag_ld(Kb, HD_, l16,      dc * 32, hf);
        v16bf kf1 = frag_ld(Kb, HD_, 16 + l16, dc * 32, hf);
        s0 = wmma_bf16(qf[dc], kf0, s0);
        s1 = wmma_bf16(qf[dc], kf1, s1);
      }
      const int kc0 = kb + l16;
      const int kc1 = kb + 16 + l16;
#pragma unroll
      for (int i = 0; i < 8; ++i) {
        const int qi = q0 + i + hf * 8;
        float v0 = s0[i] * SCALE_; if (kc0 > qi) v0 = -1e30f;
        float v1 = s1[i] * SCALE_; if (kc1 > qi) v1 = -1e30f;
        float tmax = fmaxf(v0, v1);
#pragma unroll
        for (int m = 8; m >= 1; m >>= 1) tmax = fmaxf(tmax, __shfl_xor(tmax, m, 32));
        const float mnew  = fmaxf(mrow[i], tmax);
        const float alpha = __expf(mrow[i] - mnew);
        const float p0 = __expf(v0 - mnew);
        const float p1 = __expf(v1 - mnew);
        float rs = p0 + p1;
#pragma unroll
        for (int m = 8; m >= 1; m >>= 1) rs += __shfl_xor(rs, m, 32);
        lrow[i] = lrow[i] * alpha + rs;
        mrow[i] = mnew;
#pragma unroll
        for (int dc = 0; dc < 8; ++dc) o[dc][i] *= alpha;
        const int mr = i + hf * 8;
        plds[wave][mr][l16]      = (bf16)p0;
        plds[wave][mr][16 + l16] = (bf16)p1;
      }
      // re-fragment P (C layout -> A layout) through per-wave LDS
      v16bf pf = frag_ld(&plds[wave][0][0], 32, l16, 0, hf);
#pragma unroll
      for (int dc = 0; dc < 8; ++dc) {
        v16bf vf = frag_ld(Vb, 32, dc * 16 + l16, 0, hf);
        o[dc] = wmma_bf16(pf, vf, o[dc]);
      }
    }
    ibuf ^= 1;
  }

#pragma unroll
  for (int dc = 0; dc < 8; ++dc) {
#pragma unroll
    for (int i = 0; i < 8; ++i) {
      const int mr   = i + hf * 8;
      const int grow = b * S_ + q0 + mr;
      const int col  = head * HD_ + dc * 16 + l16;
      out[(size_t)grow * (NH_ * HD_) + col] = (bf16)(o[dc][i] / lrow[i]);
    }
  }
}

// ---- output projection: [B*S,2048] x Wo^T + bo -> fp32 ---------------------
// grid (B*S/64, 8), block 256. Double-buffered async LDS A tile; wave owns
// 2 col tiles across the block's 4 M-tiles: 8 WMMAs per K-step.
__global__ __launch_bounds__(256) void oproj_kernel(
    const bf16* __restrict__ attn,   // [B*S][NH*HD]
    const bf16* __restrict__ Wot,    // [H][NH*HD] (transposed Wo)
    const float* __restrict__ bo,
    float* __restrict__ out)         // [B*S][H]
{
  __shared__ bf16 Alds[2][64 * 32];

  const int tid  = threadIdx.x;
  const int wave = tid >> 5;
  const int lane = tid & 31;
  const int hf   = lane >> 4;
  const int l16  = lane & 15;

  const int row0b = blockIdx.x * 64;
  const int cpair = blockIdx.y * 8 + wave;       // 0..63
  const int col0  = cpair * 32;
  const int col1  = col0 + 16;

  const int srow = tid >> 2;
  const int sk16 = tid & 3;
  const unsigned ldsA0 = lds_off_u32(&Alds[0][0]) + tid * 16;
  const unsigned ldsA1 = lds_off_u32(&Alds[1][0]) + tid * 16;
  const bf16*    Abase = attn + (size_t)row0b * H_;
  const unsigned vbase = (unsigned)(srow * (H_ * 2) + sk16 * 16);

  v8f c[8];
#pragma unroll
  for (int i = 0; i < 8; ++i) c[i] = (v8f){};

  async_ld_b128(ldsA0, vbase, Abase);            // prologue: tile 0
  int ibuf = 0;
  for (int k0 = 0; k0 < H_; k0 += 32) {
    wait_async0();
    __syncthreads();
    if (k0 + 32 < H_)
      async_ld_b128(ibuf ? ldsA0 : ldsA1, vbase + (unsigned)((k0 + 32) * 2), Abase);

    v16bf b0 = frag_ld(Wot, H_, col0 + l16, k0, hf);
    v16bf b1 = frag_ld(Wot, H_, col1 + l16, k0, hf);
    __builtin_prefetch(Wot + (size_t)(col0 + l16) * H_ + k0 + 64, 0, 1);
    const bf16* Ab = &Alds[ibuf][0];
#pragma unroll
    for (int mt = 0; mt < 4; ++mt) {
      v16bf a = frag_ld(Ab, 32, mt * 16 + l16, 0, hf);
      c[mt * 2]     = wmma_bf16(a, b0, c[mt * 2]);
      c[mt * 2 + 1] = wmma_bf16(a, b1, c[mt * 2 + 1]);
    }
    ibuf ^= 1;
  }

  const float bias0 = bo[col0 + l16];
  const float bias1 = bo[col1 + l16];
#pragma unroll
  for (int mt = 0; mt < 4; ++mt) {
#pragma unroll
    for (int i = 0; i < 8; ++i) {
      const int grow = row0b + mt * 16 + i + hf * 8;
      out[(size_t)grow * H_ + col0 + l16] = c[mt * 2][i] + bias0;
      out[(size_t)grow * H_ + col1 + l16] = c[mt * 2 + 1][i] + bias1;
    }
  }
}

// ---- host launch ------------------------------------------------------------
extern "C" void kernel_launch(void* const* d_in, const int* in_sizes, int n_in,
                              void* d_out, int out_size, void* d_ws, size_t ws_size,
                              hipStream_t stream) {
  (void)in_sizes; (void)n_in; (void)out_size; (void)ws_size;
  const float* hidden = (const float*)d_in[0];
  // d_in[1] = mask (strict causal; applied analytically, not read)
  const float* Wq = (const float*)d_in[2];
  const float* bq = (const float*)d_in[3];
  const float* Wk = (const float*)d_in[4];
  const float* bk = (const float*)d_in[5];
  const float* Wv = (const float*)d_in[6];
  const float* bv = (const float*)d_in[7];
  const float* Wo = (const float*)d_in[8];
  const float* bo = (const float*)d_in[9];
  float* out = (float*)d_out;

  // workspace layout (bf16 elements); attn aliases hidden_bf16 (disjoint in time)
  bf16* ws = (bf16*)d_ws;
  const size_t HB = (size_t)B_ * S_ * H_;        // 8,388,608
  const size_t W  = (size_t)H_ * NH_ * HD_;      // 4,194,304
  bf16* hiddenb = ws;                            // [B*S][H]   (later: attn out)
  bf16* wqt = ws + HB;
  bf16* wkt = wqt + W;
  bf16* wvt = wkt + W;
  bf16* wot = wvt + W;
  bf16* qb  = wot + W;                           // [B][NH][S][HD]
  bf16* kb  = qb + HB;
  bf16* vt  = kb + HB;                           // [B][NH][HD][S]
  bf16* attn = hiddenb;                          // alias

  const int thr = 256;
  const int nW  = (int)W;
  cvt_bf16_t<<<(nW + thr - 1) / thr, thr, 0, stream>>>(Wq, wqt, H_, NH_ * HD_);
  cvt_bf16_t<<<(nW + thr - 1) / thr, thr, 0, stream>>>(Wk, wkt, H_, NH_ * HD_);
  cvt_bf16_t<<<(nW + thr - 1) / thr, thr, 0, stream>>>(Wv, wvt, H_, NH_ * HD_);
  cvt_bf16_t<<<(nW + thr - 1) / thr, thr, 0, stream>>>(Wo, wot, NH_ * HD_, H_);
  cvt_bf16<<<((int)HB + thr - 1) / thr, thr, 0, stream>>>(hidden, hiddenb, (int)HB);

  qkv_rope_kernel<<<dim3(B_ * S_ / 64, 8, 3), thr, 0, stream>>>(
      hiddenb, wqt, wkt, wvt, bq, bk, bv, qb, kb, vt);

  flash_attn_kernel<<<dim3(S_ / 128, B_ * NH_), thr, 0, stream>>>(qb, kb, vt, attn);

  oproj_kernel<<<dim3(B_ * S_ / 64, 8), thr, 0, stream>>>(attn, wot, bo, out);
}
